// ResidualBlock_30812095381961
// MI455X (gfx1250) — matrix operand order repaired and checked by reading, not compiled, not run on hardware
//
#include <hip/hip_runtime.h>
#include <hip/hip_bf16.h>
#include <math.h>

// ---------------- problem constants ----------------
#define LEN   2048
#define DMODEL 768
#define DIN   1536
#define NSTATE 16
#define DTRANK 48
#define DTRANK_P 64          // padded to WMMA-K multiple
#define KCONV 4
#define NPROJ 80             // DT_RANK + 2N
#define CHUNK 128            // scan chunk length
#define NCHUNK (LEN / CHUNK) // 16

typedef __attribute__((ext_vector_type(16))) _Float16 v16h;
typedef __attribute__((ext_vector_type(8)))  _Float16 h8;
typedef __attribute__((ext_vector_type(8)))  float    v8f;

// ---------------------------------------------------------------------------
// Fragment loaders (CDNA5 ISA 7.12.2 layouts).
//   A lane(m=lane&15, h=lane>>4): halves K = [8h..8h+7] and [16+8h..16+8h+7]
//   B lane(n=lane&15, h=lane>>4): halves K = [8h..8h+7] and [16+8h..16+8h+7] of column n
// ---------------------------------------------------------------------------
__device__ __forceinline__ v16h load_frag_a(const _Float16* __restrict__ Arow,
                                            int k0, int h) {
  h8 lo = *(const h8*)(Arow + k0 + 8 * h);
  h8 hi = *(const h8*)(Arow + k0 + 16 + 8 * h);
  v16h a;
#pragma unroll
  for (int i = 0; i < 8; ++i) { a[i] = lo[i]; a[8 + i] = hi[i]; }
  return a;
}
__device__ __forceinline__ v16h load_frag_b(const _Float16* __restrict__ Brow,
                                            int k0, int h) {
  h8 lo = *(const h8*)(Brow + k0 + 8 * h);
  h8 hi = *(const h8*)(Brow + k0 + 16 + 8 * h);
  v16h b;
#pragma unroll
  for (int i = 0; i < 8; ++i) { b[i] = lo[i]; b[8 + i] = hi[i]; }
  return b;
}
__device__ __forceinline__ v8f wmma(v16h a, v16h b, v8f c) {
  return __builtin_amdgcn_wmma_f32_16x16x32_f16(false, a, false, b, (short)0, c,
                                                false, false);
}

// ---------------------------------------------------------------------------
// 32x64 tile per wave (compile-time K -> full unroll). Each k-step: 12 b128
// loads (fresh SSA temps, grouped) then 8 WMMAs; the scheduler pipelines
// next-step loads under the matrix ops within the VGPR budget.
// ---------------------------------------------------------------------------
template <int K>
__device__ __forceinline__ void wmma_tile_2x4(const _Float16* __restrict__ A,
                                              const _Float16* __restrict__ BT,
                                              int m0, int n0, int lane,
                                              v8f (&acc)[8]) {
  const int mr = lane & 15;
  const int h  = lane >> 4;
  const _Float16* Ar0 = A + (size_t)(m0 + mr) * K;
  const _Float16* Ar1 = Ar0 + (size_t)16 * K;
  const _Float16* Br  = BT + (size_t)(n0 + mr) * K;

#pragma unroll
  for (int k0 = 0; k0 < K; k0 += 32) {
    v16h a0 = load_frag_a(Ar0, k0, h);
    v16h a1 = load_frag_a(Ar1, k0, h);
    v16h b0 = load_frag_b(Br + 0 * 16 * K, k0, h);
    v16h b1 = load_frag_b(Br + 1 * 16 * K, k0, h);
    v16h b2 = load_frag_b(Br + 2 * 16 * K, k0, h);
    v16h b3 = load_frag_b(Br + 3 * 16 * K, k0, h);
    acc[0] = wmma(a0, b0, acc[0]);
    acc[1] = wmma(a0, b1, acc[1]);
    acc[2] = wmma(a0, b2, acc[2]);
    acc[3] = wmma(a0, b3, acc[3]);
    acc[4] = wmma(a1, b0, acc[4]);
    acc[5] = wmma(a1, b1, acc[5]);
    acc[6] = wmma(a1, b2, acc[6]);
    acc[7] = wmma(a1, b3, acc[7]);
  }
}

// single 16x16 tile (used by the tiny N=80 GEMM)
__device__ __forceinline__ v8f wmma_tile(const _Float16* __restrict__ A,
                                         const _Float16* __restrict__ BT,
                                         int K, int m0, int n0, int lane) {
  const int mr = lane & 15;
  const int h  = lane >> 4;
  const _Float16* Arow = A  + (size_t)(m0 + mr) * K;
  const _Float16* Brow = BT + (size_t)(n0 + mr) * K;
  v8f acc = {};
  for (int k0 = 0; k0 < K; k0 += 32) {
    v16h a = load_frag_a(Arow, k0, h);
    v16h b = load_frag_b(Brow, k0, h);
    acc = wmma(a, b, acc);
  }
  return acc;
}

// ---------------- kernel 1: RMSNorm -> f16 ----------------
__global__ void k_rmsnorm(const float* __restrict__ x,
                          const float* __restrict__ w,
                          _Float16* __restrict__ xh) {
  const int l = blockIdx.x, t = threadIdx.x;
  __shared__ float red[256];
  float ss = 0.f;
  for (int c = t; c < DMODEL; c += 256) { float v = x[l * DMODEL + c]; ss += v * v; }
  red[t] = ss; __syncthreads();
  for (int s = 128; s > 0; s >>= 1) { if (t < s) red[t] += red[t + s]; __syncthreads(); }
  const float rs = rsqrtf(red[0] / (float)DMODEL + 1e-5f);
  for (int c = t; c < DMODEL; c += 256)
    xh[l * DMODEL + c] = (_Float16)(x[l * DMODEL + c] * rs * w[c]);
}

// ------------- weight transpose f32(KxN) -> f16(N x Kp), zero-pad K -------------
__global__ void k_transpose(const float* __restrict__ src, _Float16* __restrict__ dst,
                            int K, int N, int Kp) {
  int idx = blockIdx.x * 256 + threadIdx.x;
  if (idx >= N * Kp) return;
  int n = idx / Kp, k = idx % Kp;
  dst[idx] = (k < K) ? (_Float16)src[(size_t)k * N + n] : (_Float16)0.f;
}

// ---------------- GEMM 1: xh @ W_in -> xr (xs|res) f16 ----------------
__global__ void __launch_bounds__(128, 1)
k_gemm_xr(const _Float16* __restrict__ A, const _Float16* __restrict__ BT,
          _Float16* __restrict__ xr) {
  const int lane = threadIdx.x & 31;
  const int m0 = blockIdx.x * 32;
  const int n0 = (blockIdx.y * 4 + (threadIdx.x >> 5)) * 64;
  v8f acc[8] = {};
  wmma_tile_2x4<DMODEL>(A, BT, m0, n0, lane, acc);
#pragma unroll
  for (int mt = 0; mt < 2; ++mt) {
    const int r0 = m0 + mt * 16 + 8 * (lane >> 4);
#pragma unroll
    for (int t = 0; t < 4; ++t) {
      const int col = n0 + t * 16 + (lane & 15);
#pragma unroll
      for (int r = 0; r < 8; ++r)
        xr[(size_t)(r0 + r) * (2 * DIN) + col] = (_Float16)acc[mt * 4 + t][r];
    }
  }
}

// ---------------- depthwise causal conv + SiLU -> uh f16 ----------------
__global__ void k_conv(const _Float16* __restrict__ xr, const float* __restrict__ cw,
                       const float* __restrict__ cb, _Float16* __restrict__ uh) {
  int idx = blockIdx.x * 256 + threadIdx.x;
  if (idx >= LEN * DIN) return;
  int l = idx / DIN, d = idx % DIN;
  float acc = cb[d];
#pragma unroll
  for (int k = 0; k < KCONV; ++k) {
    int t = l - (KCONV - 1) + k;
    if (t >= 0) acc += (float)xr[(size_t)t * (2 * DIN) + d] * cw[k * DIN + d];
  }
  float u = acc / (1.f + __expf(-acc));   // silu
  uh[idx] = (_Float16)u;
}

// ---------------- GEMM 2: uh @ W_x -> x_dbl f32 (+ padded delta-raw f16) --------
__global__ void k_gemm_xdbl(const _Float16* __restrict__ A, const _Float16* __restrict__ BT,
                            float* __restrict__ xdbl, _Float16* __restrict__ drh) {
  const int lane = threadIdx.x & 31;
  const int mtile = blockIdx.x;
  const int ntile = blockIdx.y * 4 + (threadIdx.x >> 5);
  if (ntile >= NPROJ / 16) return;                       // wave-uniform
  v8f acc = wmma_tile(A, BT, DIN, mtile * 16, ntile * 16, lane);
  const int col = ntile * 16 + (lane & 15);
  const int r0  = mtile * 16 + 8 * (lane >> 4);
#pragma unroll
  for (int r = 0; r < 8; ++r) {
    float v = acc[r];
    xdbl[(size_t)(r0 + r) * NPROJ + col] = v;
    if (col < DTRANK_P)
      drh[(size_t)(r0 + r) * DTRANK_P + col] =
          (col < DTRANK) ? (_Float16)v : (_Float16)0.f;
  }
}

// ---------------- GEMM 2b: delta_raw @ W_dt + b_dt, softplus -> delta f32 -------
__global__ void __launch_bounds__(128, 1)
k_gemm_delta(const _Float16* __restrict__ A, const _Float16* __restrict__ BT,
             const float* __restrict__ bdt, float* __restrict__ delta) {
  const int lane = threadIdx.x & 31;
  const int m0 = blockIdx.x * 32;
  const int n0 = (blockIdx.y * 4 + (threadIdx.x >> 5)) * 64;
  v8f acc[8] = {};
  wmma_tile_2x4<DTRANK_P>(A, BT, m0, n0, lane, acc);
#pragma unroll
  for (int mt = 0; mt < 2; ++mt) {
    const int r0 = m0 + mt * 16 + 8 * (lane >> 4);
#pragma unroll
    for (int t = 0; t < 4; ++t) {
      const int col = n0 + t * 16 + (lane & 15);
      const float b = bdt[col];
#pragma unroll
      for (int r = 0; r < 8; ++r) {
        float v = acc[mt * 4 + t][r] + b;
        float sp = (v > 20.f) ? v : log1pf(__expf(v));   // softplus
        delta[(size_t)(r0 + r) * DIN + col] = sp;
      }
    }
  }
}

// ---------------- chunked parallel scan --------------------------------------
// Lane map (wave32): lanes 0..15 -> d0, n=lane; lanes 16..31 -> d0+1, n=lane-16.
// Pass A: per (chunk, d, n) compute (P = prod dA over chunk, S = chunk-local state).
__global__ void k_scan_partial(const float* __restrict__ delta, const _Float16* __restrict__ uh,
                               const float* __restrict__ xdbl, const float* __restrict__ A_log,
                               float2* __restrict__ ps) {
  const int wid = blockIdx.x * (blockDim.x >> 5) + (threadIdx.x >> 5);
  const int lane = threadIdx.x & 31;
  const int c  = wid / (DIN / 2);         // chunk 0..NCHUNK-2
  const int dp = wid % (DIN / 2);
  const int d = 2 * dp + (lane >> 4);
  const int n = lane & 15;
  const float An = -__expf(A_log[d * NSTATE + n]);
  float P = 1.f, S = 0.f;
  const int l0 = c * CHUNK;
  for (int i = 0; i < CHUNK; ++i) {
    const int l = l0 + i;
    const float dlt = delta[(size_t)l * DIN + d];
    const float uu  = (float)uh[(size_t)l * DIN + d];
    const float Bn  = xdbl[(size_t)l * NPROJ + DTRANK + n];
    const float dA  = __expf(An * dlt);
    P *= dA;
    S = dA * S + dlt * Bn * uu;
  }
  ps[((size_t)c * DIN + d) * NSTATE + n] = make_float2(P, S);
}

// Pass B: fold preceding chunk summaries into entry state, replay chunk,
// reduce over n, apply +u*Dp and *silu(res), write f16 y.
__global__ void k_scan_final(const float* __restrict__ delta, const _Float16* __restrict__ uh,
                             const float* __restrict__ xdbl, const _Float16* __restrict__ xr,
                             const float* __restrict__ A_log, const float* __restrict__ Dp,
                             const float2* __restrict__ ps, _Float16* __restrict__ yh) {
  const int wid = blockIdx.x * (blockDim.x >> 5) + (threadIdx.x >> 5);
  const int lane = threadIdx.x & 31;
  const int c  = wid / (DIN / 2);         // chunk 0..NCHUNK-1
  const int dp = wid % (DIN / 2);
  const int d = 2 * dp + (lane >> 4);
  const int n = lane & 15;
  const float An = -__expf(A_log[d * NSTATE + n]);
  const float dcoef = Dp[d];
  // entry state = fold of chunk summaries 0..c-1
  float s = 0.f;
  for (int j = 0; j < c; ++j) {
    float2 v = ps[((size_t)j * DIN + d) * NSTATE + n];
    s = v.x * s + v.y;
  }
  const int l0 = c * CHUNK;
  for (int i = 0; i < CHUNK; ++i) {
    const int l = l0 + i;
    const float dlt = delta[(size_t)l * DIN + d];
    const float uu  = (float)uh[(size_t)l * DIN + d];
    const float Bn  = xdbl[(size_t)l * NPROJ + DTRANK + n];
    const float Cn  = xdbl[(size_t)l * NPROJ + DTRANK + NSTATE + n];
    const float dA  = __expf(An * dlt);
    s = dA * s + dlt * Bn * uu;
    float contrib = s * Cn;
#pragma unroll
    for (int off = 1; off < 16; off <<= 1)
      contrib += __shfl_xor(contrib, off, 32);
    if (n == 0) {
      float res = (float)xr[(size_t)l * (2 * DIN) + DIN + d];
      float sres = res / (1.f + __expf(-res));           // silu(res)
      float y = (contrib + uu * dcoef) * sres;
      yh[(size_t)l * DIN + d] = (_Float16)y;
    }
  }
}

// ---------------- GEMM 3: yh @ W_out + x -> out f32 ----------------
__global__ void __launch_bounds__(128, 1)
k_gemm_out(const _Float16* __restrict__ A, const _Float16* __restrict__ BT,
           const float* __restrict__ xin, float* __restrict__ out) {
  const int lane = threadIdx.x & 31;
  const int m0 = blockIdx.x * 32;
  const int n0 = (blockIdx.y * 4 + (threadIdx.x >> 5)) * 64;
  v8f acc[8] = {};
  wmma_tile_2x4<DIN>(A, BT, m0, n0, lane, acc);
#pragma unroll
  for (int mt = 0; mt < 2; ++mt) {
    const int r0 = m0 + mt * 16 + 8 * (lane >> 4);
#pragma unroll
    for (int t = 0; t < 4; ++t) {
      const int col = n0 + t * 16 + (lane & 15);
#pragma unroll
      for (int r = 0; r < 8; ++r) {
        size_t o = (size_t)(r0 + r) * DMODEL + col;
        out[o] = acc[mt * 4 + t][r] + xin[o];
      }
    }
  }
}

// ---------------------------------------------------------------------------
extern "C" void kernel_launch(void* const* d_in, const int* in_sizes, int n_in,
                              void* d_out, int out_size, void* d_ws, size_t ws_size,
                              hipStream_t stream) {
  const float* x      = (const float*)d_in[0];
  const float* norm_w = (const float*)d_in[1];
  const float* W_in   = (const float*)d_in[2];
  const float* conv_w = (const float*)d_in[3];
  const float* conv_b = (const float*)d_in[4];
  const float* W_x    = (const float*)d_in[5];
  const float* W_dt   = (const float*)d_in[6];
  const float* b_dt   = (const float*)d_in[7];
  const float* A_log  = (const float*)d_in[8];
  const float* Dp     = (const float*)d_in[9];
  const float* W_out  = (const float*)d_in[10];
  float* out = (float*)d_out;

  // ---- workspace layout (f16 activations + transposed f16 weights) ----
  char* p = (char*)d_ws;
  size_t off = 0;
  auto take = [&](size_t bytes) { void* q = p + off; off = (off + bytes + 255) & ~(size_t)255; return q; };
  _Float16* xh    = (_Float16*)take((size_t)LEN * DMODEL * 2);       // normed x
  _Float16* WinT  = (_Float16*)take((size_t)(2 * DIN) * DMODEL * 2); // (3072 x 768)
  _Float16* xr    = (_Float16*)take((size_t)LEN * 2 * DIN * 2);      // xs|res
  _Float16* uh    = (_Float16*)take((size_t)LEN * DIN * 2);
  _Float16* WxT   = (_Float16*)take((size_t)NPROJ * DIN * 2);        // (80 x 1536)
  float*    xdbl  = (float*)   take((size_t)LEN * NPROJ * 4);
  _Float16* drh   = (_Float16*)take((size_t)LEN * DTRANK_P * 2);     // padded delta-raw
  _Float16* WdtT  = (_Float16*)take((size_t)DIN * DTRANK_P * 2);     // (1536 x 64)
  float*    delta = (float*)   take((size_t)LEN * DIN * 4);
  _Float16* yh    = (_Float16*)take((size_t)LEN * DIN * 2);
  _Float16* WoutT = (_Float16*)take((size_t)DMODEL * DIN * 2);       // (768 x 1536)
  float2*   ps    = (float2*)  take((size_t)(NCHUNK - 1) * DIN * NSTATE * 8);
  (void)ws_size; (void)in_sizes; (void)n_in; (void)out_size;

  // 1) RMSNorm
  k_rmsnorm<<<LEN, 256, 0, stream>>>(x, norm_w, xh);

  // 2) weight transposes (f32 KxN -> f16 NxKp)
  k_transpose<<<(2 * DIN * DMODEL + 255) / 256, 256, 0, stream>>>(W_in,  WinT,  DMODEL, 2 * DIN, DMODEL);
  k_transpose<<<(NPROJ * DIN + 255) / 256,      256, 0, stream>>>(W_x,   WxT,   DIN,    NPROJ,   DIN);
  k_transpose<<<(DIN * DTRANK_P + 255) / 256,   256, 0, stream>>>(W_dt,  WdtT,  DTRANK, DIN,     DTRANK_P);
  k_transpose<<<(DMODEL * DIN + 255) / 256,     256, 0, stream>>>(W_out, WoutT, DIN,    DMODEL,  DIN);

  // 3) GEMM1: (2048x768)x(768x3072) -> xr   (32x64 tiles)
  k_gemm_xr<<<dim3(LEN / 32, (2 * DIN / 64) / 4), 128, 0, stream>>>(xh, WinT, xr);

  // 4) depthwise conv + SiLU -> uh
  k_conv<<<(LEN * DIN + 255) / 256, 256, 0, stream>>>(xr, conv_w, conv_b, uh);

  // 5) GEMM2: (2048x1536)x(1536x80) -> xdbl (+ padded delta-raw f16)
  k_gemm_xdbl<<<dim3(LEN / 16, 2), 128, 0, stream>>>(uh, WxT, xdbl, drh);

  // 6) GEMM2b: (2048x64)x(64x1536) + b_dt, softplus -> delta   (32x64 tiles)
  k_gemm_delta<<<dim3(LEN / 32, (DIN / 64) / 4), 128, 0, stream>>>(drh, WdtT, b_dt, delta);

  // 7) chunked parallel scan: partials, then finalize with gating epilogue
  k_scan_partial<<<(NCHUNK - 1) * (DIN / 2) / 8, 256, 0, stream>>>(delta, uh, xdbl, A_log, ps);
  k_scan_final<<<NCHUNK * (DIN / 2) / 8, 256, 0, stream>>>(delta, uh, xdbl, xr, A_log, Dp, ps, yh);

  // 8) GEMM3: (2048x1536)x(1536x768) + residual -> out   (32x64 tiles)
  k_gemm_out<<<dim3(LEN / 32, (DMODEL / 64) / 4), 128, 0, stream>>>(yh, WoutT, x, out);
}
